// RMILoss_22454089023997
// MI455X (gfx1250) — matrix operand order repaired and checked by reading, not compiled, not run on hardware
//
#include <hip/hip_runtime.h>
#include <hip/hip_bf16.h>

typedef __attribute__((ext_vector_type(16))) _Float16 v16h;
typedef __attribute__((ext_vector_type(8)))  float    v8f;
typedef __attribute__((ext_vector_type(4)))  unsigned int u32x4;
typedef __attribute__((ext_vector_type(8)))  int      i32x8;
typedef __attribute__((ext_vector_type(4)))  int      i32x4;

#define NUM_CLASSES 21
#define N_BATCH 4
#define H 512
#define W 512
#define HW (H * W)
#define PH 171                 // pooled H (k=s=3, pad=1)
#define PW 171
#define PHW (PH * PW)          // 29241
#define NH 169                 // shifted-crop dims
#define NW 169
#define LCOLS (NH * NW)        // 28561 columns per Gram
#define NC (N_BATCH * NUM_CLASSES)   // 84 (n,c) pairs
#define ALPHA 0.0005f
#define KCHUNKS ((LCOLS + 31) / 32)  // 893

// ---- workspace layout (bytes) ----
#define CE_OFF      0                                   // 2 floats: nll sum, valid count
#define GRAM_OFF    256
#define GRAM_FLOATS (NC * 3 * 256)                      // 84 x {ll,pp,lp} x 16x16 f32
#define ROWSUM_OFF  (GRAM_OFF + GRAM_FLOATS * 4)
#define ROWSUM_FLOATS (NC * 32)
#define LA_OFF      ((ROWSUM_OFF + ROWSUM_FLOATS * 4 + 255) & ~(size_t)255)
#define POOL_HALFS  ((size_t)NC * PHW)
#define PR_OFF      ((LA_OFF + POOL_HALFS * 2 + 255) & ~(size_t)255)

// ---------------------------------------------------------------------------
__global__ void zero_kernel(float* ws, int n) {
    int i = blockIdx.x * blockDim.x + threadIdx.x;
    if (i < n) ws[i] = 0.0f;
}

// ---------------------------------------------------------------------------
// Cross entropy with ignore_index=255.
__global__ void ce_kernel(const float* __restrict__ score,
                          const int* __restrict__ target,
                          float* __restrict__ accum) {
    int p = blockIdx.x * blockDim.x + threadIdx.x;
    float nll = 0.0f, cnt = 0.0f;
    if (p < N_BATCH * HW) {
        int n = p / HW, hw = p - n * HW;
        const float* s = score + (size_t)n * NUM_CLASSES * HW + hw;
        float mx = -1e30f;
        #pragma unroll
        for (int c = 0; c < NUM_CLASSES; ++c) mx = fmaxf(mx, s[(size_t)c * HW]);
        float se = 0.0f;
        #pragma unroll
        for (int c = 0; c < NUM_CLASSES; ++c) se += __expf(s[(size_t)c * HW] - mx);
        int t = target[p];
        if (t != 255) {
            cnt = 1.0f;
            int ts = (t >= 0 && t < NUM_CLASSES) ? t : 0;
            nll = -(s[(size_t)ts * HW] - mx - __logf(se));
        }
    }
    #pragma unroll
    for (int off = 16; off > 0; off >>= 1) {
        nll += __shfl_down(nll, off);
        cnt += __shfl_down(cnt, off);
    }
    if ((threadIdx.x & 31) == 0) {
        atomicAdd(&accum[0], nll);
        atomicAdd(&accum[1], cnt);
    }
}

// ---------------------------------------------------------------------------
// Fused one-hot/sigmoid/clip + 3x3 s3 max pool (pad 1), stored fp16.
__global__ void pool_kernel(const float* __restrict__ score,
                            const int* __restrict__ target,
                            _Float16* __restrict__ la_pool,
                            _Float16* __restrict__ pr_pool) {
    int q = blockIdx.x * blockDim.x + threadIdx.x;
    if (q >= NC * PHW) return;
    int nc = q / PHW, r = q - nc * PHW;
    int n = nc / NUM_CLASSES, c = nc - n * NUM_CLASSES;
    int py = r / PW, px = r - py * PW;
    float lam = 0.0f, prm = 0.0f;
    #pragma unroll
    for (int dy = -1; dy <= 1; ++dy) {
        int y = py * 3 + dy;
        if ((unsigned)y >= (unsigned)H) continue;
        #pragma unroll
        for (int dx = -1; dx <= 1; ++dx) {
            int x = px * 3 + dx;
            if ((unsigned)x >= (unsigned)W) continue;
            int t = target[(size_t)n * HW + y * W + x];
            float s = score[((size_t)n * NUM_CLASSES + c) * HW + y * W + x];
            bool mask = (t >= 0 && t < NUM_CLASSES);
            float la = (mask && t == c) ? 1.0f : 0.0f;
            float sig = 1.0f / (1.0f + __expf(-s));
            float pr = fminf(fmaxf(mask ? sig : 0.0f, 1e-6f), 1.0f);
            lam = fmaxf(lam, la);
            prm = fmaxf(prm, pr);
        }
    }
    la_pool[q] = (_Float16)lam;
    pr_pool[q] = (_Float16)prm;
}

// ---------------------------------------------------------------------------
// Row sums of the 9 x LCOLS shifted-crop matrices.
__global__ void rowsum_kernel(const _Float16* __restrict__ la_pool,
                              const _Float16* __restrict__ pr_pool,
                              float* __restrict__ rowsum) {
    int nc = blockIdx.x;
    const _Float16* la = la_pool + (size_t)nc * PHW;
    const _Float16* pr = pr_pool + (size_t)nc * PHW;
    float sl[9], sp[9];
    #pragma unroll
    for (int d = 0; d < 9; ++d) { sl[d] = 0.0f; sp[d] = 0.0f; }
    for (int col = threadIdx.x; col < LCOLS; col += blockDim.x) {
        int y = col / NW, x = col - y * NW;
        #pragma unroll
        for (int d = 0; d < 9; ++d) {
            int idx = (y + d / 3) * PW + (x + d % 3);
            sl[d] += (float)la[idx];
            sp[d] += (float)pr[idx];
        }
    }
    __shared__ float red[256];
    for (int d = 0; d < 9; ++d) {
        red[threadIdx.x] = sl[d];
        __syncthreads();
        for (int s = 128; s > 0; s >>= 1) {
            if ((int)threadIdx.x < s) red[threadIdx.x] += red[threadIdx.x + s];
            __syncthreads();
        }
        if (threadIdx.x == 0) rowsum[nc * 32 + d] = red[0];
        __syncthreads();
        red[threadIdx.x] = sp[d];
        __syncthreads();
        for (int s = 128; s > 0; s >>= 1) {
            if ((int)threadIdx.x < s) red[threadIdx.x] += red[threadIdx.x + s];
            __syncthreads();
        }
        if (threadIdx.x == 0) rowsum[nc * 32 + 16 + d] = red[0];
        __syncthreads();
    }
}

// ---------------------------------------------------------------------------
// TDM: 1-D tile copy global -> LDS, D# per CDNA5 ISA.
// group0: count=1 | lds_addr[63:32] | global_addr[120:64] | type=2 (127:126)
// group1: data_size=1 (2B) | tensor_dim0=n | tensor_dim1=1 | tile_dim0=n |
//         tile_dim1=1 | tensor_dim0_stride=n
// 6-arg builtin form (clang-23 / therock): (g0, g1, g2, g3, g4, cpol)
__device__ __forceinline__ void tdm_load_1d(const void* gptr, unsigned int lds_addr,
                                            unsigned int n) {
    unsigned long long ga = (unsigned long long)(uintptr_t)gptr;
    u32x4 g0;
    g0[0] = 1u;                                        // count=1
    g0[1] = lds_addr;                                  // LDS byte address
    g0[2] = (unsigned int)(ga & 0xffffffffu);          // global_addr[31:0]
    g0[3] = (unsigned int)((ga >> 32) & 0x1ffffffu) | (2u << 30); // addr[56:32], type=2
    i32x8 g1;
    g1[0] = (int)(1u << 16);                           // wg_mask=0, data_size=2B
    g1[1] = (int)((n & 0xffffu) << 16);                // tensor_dim0[15:0]
    g1[2] = (int)(((n >> 16) & 0xffffu) | (1u << 16)); // tensor_dim0[31:16], tensor_dim1=1
    g1[3] = (int)((n & 0xffffu) << 16);                // tile_dim0 = n
    g1[4] = 1;                                         // tile_dim1=1, tile_dim2=0
    g1[5] = (int)n;                                    // tensor_dim0_stride[31:0]
    g1[6] = 0;                                         // stride hi, dim1_stride lo
    g1[7] = 0;
    i32x4 z4 = {0, 0, 0, 0};
    i32x8 z8 = {0, 0, 0, 0, 0, 0, 0, 0};
    __builtin_amdgcn_tensor_load_to_lds(g0, g1, z4, z4, z8, 0);
}

// Branchless fetch of X[d][col] from the LDS-staged pooled map.
__device__ __forceinline__ _Float16 lds_fetch(const _Float16* t, int base, int col,
                                              bool vrow) {
    int y = col / NW;
    int x = col - y * NW;
    _Float16 v = t[base + y * PW + x];                 // unconditional, padded region
    bool ok = vrow && (col < LCOLS);
    return ok ? v : (_Float16)0.0f;
}

// Gram kernel: one block (8 waves) per (n,c). Stage la/pr pooled maps into LDS
// with TENSOR_LOAD_TO_LDS, then accumulate la*la^T, pr*pr^T, la*pr^T over
// K=28561 in chunks of 32 with v_wmma_f32_16x16x32_f16 (f32 accumulate).
__global__ void __launch_bounds__(256) gram_kernel(const _Float16* __restrict__ la_pool,
                                                   const _Float16* __restrict__ pr_pool,
                                                   float* __restrict__ gram) {
    // la tile | pr tile | pad (fetch tail may over-read; value cndmask'ed to 0)
    __shared__ _Float16 tile[2 * PHW + 512];
    int nc = blockIdx.x;

    if (threadIdx.x < 32) {                            // wave 0 issues both DMAs
        unsigned int lds0 = (unsigned int)(uintptr_t)&tile[0];
        unsigned int lds1 = (unsigned int)(uintptr_t)&tile[PHW];
        tdm_load_1d(la_pool + (size_t)nc * PHW, lds0, PHW);
        tdm_load_1d(pr_pool + (size_t)nc * PHW, lds1, PHW);
        __builtin_amdgcn_s_wait_tensorcnt(0);
    }
    __syncthreads();

    int wave = threadIdx.x >> 5;
    int nwaves = blockDim.x >> 5;
    int lane = threadIdx.x & 31;
    int hi = lane >> 4;
    int m = lane & 15;                                 // A row / B col = crop index d
    bool vd = (m < 9);
    int base = vd ? (m / 3) * PW + (m % 3) : 0;
    int hi8 = hi ? 8 : 0;
    int hi16 = hi ? 16 : 0;
    const _Float16* sLa = tile;
    const _Float16* sPr = tile + PHW;

    v8f acc_ll = {};
    v8f acc_pp = {};
    v8f acc_lp = {};

    for (int ch = wave; ch < KCHUNKS; ch += nwaves) {
        int kbase = ch * 32;
        v16h aLa, bLa, aPr, bPr;
        #pragma unroll
        for (int j = 0; j < 16; ++j) {
            // A 16x32 fp16 layout: lanes 0-15 -> K 0-7,16-23; lanes 16-31 -> K 8-15,24-31
            int kA = kbase + j + ((j & 8) ? 8 : 0) + hi8;
            // B 32x16 fp16 layout: lanes 0-15 -> rows K 0-7,16-23; lanes 16-31 -> rows K 8-15,24-31
            int kB = kbase + j + ((j & 8) ? 8 : 0) + hi8;
            aLa[j] = lds_fetch(sLa, base, kA, vd);
            bLa[j] = lds_fetch(sLa, base, kB, vd);
            aPr[j] = lds_fetch(sPr, base, kA, vd);
            bPr[j] = lds_fetch(sPr, base, kB, vd);
        }
        acc_ll = __builtin_amdgcn_wmma_f32_16x16x32_f16(false, aLa, false, bLa,
                                                        (short)0, acc_ll, false, false);
        acc_pp = __builtin_amdgcn_wmma_f32_16x16x32_f16(false, aPr, false, bPr,
                                                        (short)0, acc_pp, false, false);
        acc_lp = __builtin_amdgcn_wmma_f32_16x16x32_f16(false, aLa, false, bPr,
                                                        (short)0, acc_lp, false, false);
    }

    // C/D 16x16 f32 layout: VGPR r, lanes 0-15 -> M=r; lanes 16-31 -> M=r+8; N=lane&15
    float* g = gram + (size_t)nc * 3 * 256;
    #pragma unroll
    for (int r2 = 0; r2 < 8; ++r2) {
        int o = (r2 + hi8) * 16 + m;
        atomicAdd(&g[o],       acc_ll[r2]);
        atomicAdd(&g[256 + o], acc_pp[r2]);
        atomicAdd(&g[512 + o], acc_lp[r2]);
    }
}

// ---------------------------------------------------------------------------
// Finalize: centered covariances, 9x9 inverse, appro_var, Cholesky log-det.
__global__ void finalize_kernel(const float* __restrict__ gram,
                                const float* __restrict__ rowsum,
                                const float* __restrict__ ce,
                                float* __restrict__ out) {
    __shared__ float red[128];
    int tid = threadIdx.x;
    float rmi = 0.0f;
    if (tid < NC) {
        const float* g = gram + (size_t)tid * 3 * 256;
        float sl[9], sp[9];
        #pragma unroll
        for (int i = 0; i < 9; ++i) {
            sl[i] = rowsum[tid * 32 + i];
            sp[i] = rowsum[tid * 32 + 16 + i];
        }
        const float invL = 1.0f / (float)LCOLS;
        float ll[9][9], pp[9][9], lp[9][9], inv[9][9];
        for (int i = 0; i < 9; ++i)
            for (int j = 0; j < 9; ++j) {
                ll[i][j] = g[i * 16 + j] - sl[i] * sl[j] * invL;
                pp[i][j] = g[256 + i * 16 + j] - sp[i] * sp[j] * invL + ((i == j) ? ALPHA : 0.0f);
                lp[i][j] = g[512 + i * 16 + j] - sl[i] * sp[j] * invL;
                inv[i][j] = (i == j) ? 1.0f : 0.0f;
            }
        for (int k = 0; k < 9; ++k) {
            float piv = 1.0f / pp[k][k];
            for (int j = 0; j < 9; ++j) { pp[k][j] *= piv; inv[k][j] *= piv; }
            for (int i = 0; i < 9; ++i) {
                if (i == k) continue;
                float f = pp[i][k];
                for (int j = 0; j < 9; ++j) { pp[i][j] -= f * pp[k][j]; inv[i][j] -= f * inv[k][j]; }
            }
        }
        float t1[9][9], Mv[9][9];
        for (int i = 0; i < 9; ++i)
            for (int j = 0; j < 9; ++j) {
                float a = 0.0f;
                for (int k = 0; k < 9; ++k) a += lp[i][k] * inv[k][j];
                t1[i][j] = a;
            }
        for (int i = 0; i < 9; ++i)
            for (int j = 0; j < 9; ++j) {
                float a = 0.0f;
                for (int k = 0; k < 9; ++k) a += t1[i][k] * lp[j][k];
                Mv[i][j] = ll[i][j] - a + ((i == j) ? ALPHA : 0.0f);
            }
        for (int k = 0; k < 9; ++k) {
            float d = Mv[k][k];
            for (int q = 0; q < k; ++q) d -= Mv[k][q] * Mv[k][q];
            d = sqrtf(fmaxf(d, 1e-20f));
            Mv[k][k] = d;
            float id = 1.0f / d;
            for (int i = k + 1; i < 9; ++i) {
                float v = Mv[i][k];
                for (int q = 0; q < k; ++q) v -= Mv[i][q] * Mv[k][q];
                Mv[i][k] = v * id;
            }
        }
        float logdet = 0.0f;
        for (int k = 0; k < 9; ++k) logdet += __logf(Mv[k][k] + 1e-8f);
        rmi = logdet;   // 0.5 * (2 * logdet)
    }
    red[tid] = rmi;
    __syncthreads();
    for (int s = 64; s > 0; s >>= 1) {
        if (tid < s) red[tid] += red[tid + s];
        __syncthreads();
    }
    if (tid == 0) {
        float rmi_loss = red[0] / (float)(N_BATCH * 9);
        float ce_loss = ce[0] / fmaxf(ce[1], 1.0f);
        out[0] = 0.5f * ce_loss + 0.5f * rmi_loss;
    }
}

// ---------------------------------------------------------------------------
extern "C" void kernel_launch(void* const* d_in, const int* in_sizes, int n_in,
                              void* d_out, int out_size, void* d_ws, size_t ws_size,
                              hipStream_t stream) {
    const float* score = (const float*)d_in[0];
    const int* target = (const int*)d_in[1];
    float* out = (float*)d_out;
    char* ws = (char*)d_ws;

    float* ce = (float*)(ws + CE_OFF);
    float* gram = (float*)(ws + GRAM_OFF);
    float* rowsum = (float*)(ws + ROWSUM_OFF);
    _Float16* la_pool = (_Float16*)(ws + LA_OFF);
    _Float16* pr_pool = (_Float16*)(ws + PR_OFF);

    int zn = GRAM_OFF / 4 + GRAM_FLOATS;
    zero_kernel<<<(zn + 255) / 256, 256, 0, stream>>>((float*)ws, zn);

    ce_kernel<<<(N_BATCH * HW + 255) / 256, 256, 0, stream>>>(score, target, ce);

    pool_kernel<<<(NC * PHW + 255) / 256, 256, 0, stream>>>(score, target, la_pool, pr_pool);

    rowsum_kernel<<<NC, 256, 0, stream>>>(la_pool, pr_pool, rowsum);

    gram_kernel<<<NC, 256, 0, stream>>>(la_pool, pr_pool, gram);

    finalize_kernel<<<1, 128, 0, stream>>>(gram, rowsum, ce, out);
}
